// QuantizationLayer_24601572671786
// MI455X (gfx1250) — matrix operand order repaired and checked by reading, not compiled, not run on hardware
//
#include <hip/hip_runtime.h>
#include <hip/hip_bf16.h>

typedef __attribute__((ext_vector_type(16))) _Float16 v16h;
typedef __attribute__((ext_vector_type(8)))  float    v8f;

#define B_    64
#define D_    256
#define HW_   1024
#define K_    1024
#define N_    65536            // B_*HW_
#define NT_   64               // K_/16 code tiles
#define KS_   8                // D_/32 k-steps
#define FPT_  (KS_ * 32)       // fragments per code tile (256)
#define TOT_  16777216         // B_*D_*HW_

// ---------------------------------------------------------------------------
// Kernel 1: pack embed [D,K] f32 into WMMA A-fragment order, f16.
// A is 16x32 (codes x d). Layout: lane l (lg=l/16, m=l%16), half h:
// d_off = lg*16 + h (contiguous, matches B fragment).
// ---------------------------------------------------------------------------
__global__ void vq_pack_embed(const float* __restrict__ embed,
                              _Float16* __restrict__ Apack) {
    int id = blockIdx.x * blockDim.x + threadIdx.x;   // 64*8*32*16 = 262144
    int h = id & 15;
    int l = (id >> 4) & 31;
    int s = (id >> 9) & 7;
    int j = id >> 12;
    int lg = l >> 4;
    int v = h >> 1, p = h & 1;
    int doff = lg * 16 + h;
    int d = s * 32 + doff;
    int code = j * 16 + (l & 15);
    Apack[id] = (_Float16)embed[d * K_ + code];
}

// ---------------------------------------------------------------------------
// Kernel 2: hn[k] = -0.5 * ||e_k||^2   (seeds the WMMA accumulator)
// ---------------------------------------------------------------------------
__global__ void vq_halfnorm(const float* __restrict__ embed,
                            float* __restrict__ hn) {
    int k = blockIdx.x * blockDim.x + threadIdx.x;    // 1024 threads
    float s = 0.f;
    for (int d = 0; d < D_; ++d) {
        float e = embed[d * K_ + k];
        s += e * e;
    }
    hn[k] = -0.5f * s;
}

// ---------------------------------------------------------------------------
// Kernel 3: transpose x [B,D,H,W] f32 -> xt [N,D] f16 (LDS 32x32 tiles).
// xt row layout makes the WMMA B fragment a contiguous 32B load per lane.
// ---------------------------------------------------------------------------
__global__ void vq_transpose_x(const float* __restrict__ x,
                               _Float16* __restrict__ xt) {
    __shared__ float tile[32][33];
    int b  = blockIdx.x;          // 64
    int dt = blockIdx.y;          // 8  (d tile)
    int wt = blockIdx.z;          // 32 (hw tile)
    int tx = threadIdx.x;         // 32
    int ty = threadIdx.y;         // 8
    const float* xb = x + (size_t)b * D_ * HW_;
#pragma unroll
    for (int i = 0; i < 4; ++i) {
        int d  = dt * 32 + ty + i * 8;
        int hw = wt * 32 + tx;
        tile[ty + i * 8][tx] = xb[d * HW_ + hw];
    }
    __syncthreads();
#pragma unroll
    for (int i = 0; i < 4; ++i) {
        int row = ty + i * 8;                  // hw within tile
        int n   = b * HW_ + wt * 32 + row;
        int d   = dt * 32 + tx;
        xt[(size_t)n * D_ + d] = (_Float16)tile[tx][row];
    }
}

// ---------------------------------------------------------------------------
// Kernel 4: fused WMMA GEMM + argmax with LDS double-buffered codebook tiles.
// 256 threads = 8 waves; each wave owns 16 points. All waves share the same
// 8KB A tile, staged in LDS (2x8KB ping-pong). Per iteration: global loads
// for tile j+1 are issued into one v16h register per thread, the whole
// tile's A fragments are preloaded from LDS into a per-iteration register
// array (so ds waits pipeline against WMMA issue instead of gating each
// WMMA), the 8-WMMA chain runs, then staged data is committed + barrier.
// C is seeded with -0.5||e||^2 so the matrix pipe emits final scores;
// two accumulators break the RAW chain.
// ---------------------------------------------------------------------------
__global__ void __launch_bounds__(256)
vq_argmax(const _Float16* __restrict__ Apack,
          const _Float16* __restrict__ xt,
          const float* __restrict__ hn,
          int* __restrict__ ind) {
    __shared__ _Float16 atile[2][FPT_ * 16];              // 2 x 8KB

    int tid  = threadIdx.x;
    int lane = tid & 31;
    int wave = tid >> 5;
    int pbase = blockIdx.x * 128 + wave * 16;
    int lg = lane >> 4;
    int pt = pbase + (lane & 15);

    const v16h* afr = (const v16h*)Apack;                 // fragment-indexed
    const _Float16* brow = xt + (size_t)pt * D_;          // this lane's point

    // B fragments: loop-invariant, keep in registers (64 VGPRs)
    v16h bfrag[KS_];
#pragma unroll
    for (int s = 0; s < KS_; ++s)
        bfrag[s] = *(const v16h*)(brow + s * 32 + lg * 16);

    // stage tile 0: each thread moves one 32B fragment
    *(v16h*)&atile[0][tid * 16] = afr[tid];
    __syncthreads();

    float best = -3.4e38f;
    int bestidx = 0;

    for (int j = 0; j < NT_; ++j) {
        // issue global loads for tile j+1 (uniform guard, EXEC stays full)
        v16h stage;
        if (j + 1 < NT_) {
            stage = afr[(j + 1) * FPT_ + tid];
            __builtin_prefetch(Apack + (size_t)((j + 2) & (NT_ - 1)) * FPT_ * 16, 0, 3);
        }

        // ---- preload the whole tile's A fragments from LDS ----
        // lives only within this iteration -> no cross-iteration pressure
        const _Float16* at = atile[j & 1];
        v16h a[KS_];
#pragma unroll
        for (int s = 0; s < KS_; ++s)
            a[s] = *(const v16h*)(at + (s * 32 + lane) * 16);

        // ---- 8-WMMA chain, two accumulators for ILP ----
        v8f c0 = *(const v8f*)(hn + j * 16 + lg * 8);
        v8f c1 = {};
#pragma unroll
        for (int s = 0; s < KS_; s += 2) {
            c0 = __builtin_amdgcn_wmma_f32_16x16x32_f16(
                    false, a[s],     false, bfrag[s],     (short)0, c0, false, false);
            c1 = __builtin_amdgcn_wmma_f32_16x16x32_f16(
                    false, a[s + 1], false, bfrag[s + 1], (short)0, c1, false, false);
        }
        int kbase = j * 16 + lg * 8;
#pragma unroll
        for (int r = 0; r < 8; ++r) {
            float v = c0[r] + c1[r];
            if (v > best) { best = v; bestidx = kbase + r; }
        }

        // commit staged tile j+1 to the other buffer; it was last read in
        // iteration j-1 before that iteration's barrier -> race-free
        if (j + 1 < NT_)
            *(v16h*)&atile[(j + 1) & 1][tid * 16] = stage;
        __syncthreads();
    }

    // merge the two half-lane code groups (same point, codes split 0-7/8-15)
    float ob = __shfl_xor(best, 16);
    int   oi = __shfl_xor(bestidx, 16);
    if (ob > best || (ob == best && oi < bestidx)) { best = ob; bestidx = oi; }
    if (lane < 16) ind[pt] = bestidx;
}

// ---------------------------------------------------------------------------
// Kernel 5: scatter embed[d, ind[n]] into both outputs in [B,D,H,W] layout.
// ---------------------------------------------------------------------------
__global__ void vq_scatter(const float* __restrict__ embed,
                           const int* __restrict__ ind,
                           float* __restrict__ out) {
    int g = blockIdx.x * blockDim.x + threadIdx.x;        // TOT_ threads
    int b  = g >> 18;             // / (D_*HW_)
    int r  = g & (D_ * HW_ - 1);
    int d  = r >> 10;
    int hw = r & (HW_ - 1);
    int idx = ind[b * HW_ + hw];
    float v = embed[d * K_ + idx];
    out[g] = v;
    out[g + TOT_] = v;
}

extern "C" void kernel_launch(void* const* d_in, const int* in_sizes, int n_in,
                              void* d_out, int out_size, void* d_ws, size_t ws_size,
                              hipStream_t stream) {
    const float* x     = (const float*)d_in[0];
    const float* embed = (const float*)d_in[1];
    float* out         = (float*)d_out;

    // workspace layout
    char* wsb = (char*)d_ws;
    const size_t XT_BYTES    = (size_t)N_ * D_ * sizeof(_Float16);   // 32 MB
    const size_t APACK_BYTES = (size_t)NT_ * FPT_ * 16 * sizeof(_Float16); // 512 KB
    _Float16* xt    = (_Float16*)(wsb);
    _Float16* Apack = (_Float16*)(wsb + XT_BYTES);
    float*    hn    = (float*)(wsb + XT_BYTES + APACK_BYTES);
    int*      ind   = (int*)(wsb + XT_BYTES + APACK_BYTES + 4096);

    // 1) pack codebook into WMMA A-fragment order (f16)
    vq_pack_embed<<<1024, 256, 0, stream>>>(embed, Apack);
    // 2) -0.5*||e_k||^2
    vq_halfnorm<<<4, 256, 0, stream>>>(embed, hn);
    // 3) transpose x -> xt [N, D] f16
    vq_transpose_x<<<dim3(B_, 8, 32), dim3(32, 8), 0, stream>>>(x, xt);
    // 4) fused WMMA distance + argmax (LDS double-buffered tiles)
    vq_argmax<<<N_ / 128, 256, 0, stream>>>(Apack, xt, hn, ind);
    // 5) gather/scatter outputs (quantize_st and diff are value-identical)
    vq_scatter<<<TOT_ / 256, 256, 0, stream>>>(embed, ind, out);
}
